// QTable_71347996721418
// MI455X (gfx1250) — matrix operand order repaired and checked by reading, not compiled, not run on hardware
//
#include <hip/hip_runtime.h>

// ---------------------------------------------------------------------------
// Problem constants (B,T,V,D) = (4,2048,8192,256)
// ---------------------------------------------------------------------------
#define BB 4
#define TT_DIM 2048
#define VV_DIM 8192
#define DD_DIM 256
#define MROWS (BB * TT_DIM)        // 8192
#define NQKV  (3 * DD_DIM)         // 768

typedef __attribute__((ext_vector_type(16))) __bf16   v16bf;
typedef __attribute__((ext_vector_type(8)))  float    v8f;
typedef __attribute__((ext_vector_type(4)))  float    v4f;
typedef __attribute__((ext_vector_type(8)))  unsigned v8u;
typedef __attribute__((ext_vector_type(4)))  unsigned v4u;
typedef __attribute__((ext_vector_type(4)))  int      v4i;

#define AS1 __attribute__((address_space(1)))
#define AS3 __attribute__((address_space(3)))

#if __has_builtin(__builtin_amdgcn_global_load_async_to_lds_b128) && \
    __has_builtin(__builtin_amdgcn_s_wait_asynccnt)
#define HAS_ASYNC_LDS 1
#else
#define HAS_ASYNC_LDS 0
#endif

// Async 16-byte global -> LDS copy (per lane), ASYNCcnt-tracked. Fallback: sync.
__device__ __attribute__((always_inline)) inline void
cp_async_b128(const unsigned short* gsrc, unsigned short* ldst) {
#if HAS_ASYNC_LDS
    __builtin_amdgcn_global_load_async_to_lds_b128(
        (AS1 v4i*)gsrc, (AS3 v4i*)ldst, 0, 0);
#else
    *(v4u*)ldst = *(const v4u*)gsrc;
#endif
}

template <int N>
__device__ __attribute__((always_inline)) inline void async_wait_n() {
#if HAS_ASYNC_LDS
    __builtin_amdgcn_s_wait_asynccnt(N);
#endif
}

__device__ __attribute__((always_inline)) inline unsigned short f2bf(float f) {
    unsigned u = __builtin_bit_cast(unsigned, f);
    unsigned r = 0x7FFFu + ((u >> 16) & 1u);
    return (unsigned short)((u + r) >> 16);
}

// A fragment: 16x32 bf16, layout [m][k] in LDS. Lane l: m = l%16, half = l/16.
// VGPR pair q holds K = (q<4?0:16) + half*8 + (q&3)*2  (contiguous k,k+1).
__device__ __attribute__((always_inline)) inline v16bf
ld_frag_a(const unsigned short* tile, int stride, int row, int half, int kofs) {
    v8u u;
    const unsigned short* p = tile + row * stride + kofs;
#pragma unroll
    for (int q = 0; q < 8; ++q) {
        int k = ((q & 4) ? 16 : 0) + half * 8 + (q & 3) * 2;
        u[q] = *(const unsigned*)(p + k);
    }
    return __builtin_bit_cast(v16bf, u);
}

// B fragment: 32x16 bf16, layout [n][k] in LDS. Lane l: n = l%16, half = l/16.
// VGPR pair q holds K = (q<4?0:16) + half*8 + (q&3)*2  (contiguous k,k+1).
__device__ __attribute__((always_inline)) inline v16bf
ld_frag_b(const unsigned short* tile, int stride, int row, int half, int kofs) {
    v8u u;
    const unsigned short* p = tile + row * stride + kofs;
#pragma unroll
    for (int q = 0; q < 8; ++q) {
        int k = ((q & 4) ? 16 : 0) + half * 8 + (q & 3) * 2;
        u[q] = *(const unsigned*)(p + k);
    }
    return __builtin_bit_cast(v16bf, u);
}

__device__ __attribute__((always_inline)) inline v8f
wmma_bf16(v16bf a, v16bf b, v8f c) {
    return __builtin_amdgcn_wmma_f32_16x16x32_bf16(
        false, a, false, b, (short)0, c, false, false);
}

// ---------------------------------------------------------------------------
// f32 -> bf16 elementwise convert (for weights)
// ---------------------------------------------------------------------------
__global__ __launch_bounds__(256) void cvt_f32_bf16(const float* __restrict__ src,
                                                    unsigned short* __restrict__ dst,
                                                    int n) {
    int i = blockIdx.x * blockDim.x + threadIdx.x;
    int stride = gridDim.x * blockDim.x;
    for (; i < n; i += stride) dst[i] = f2bf(src[i]);
}

// ---------------------------------------------------------------------------
// Tiled GEMM:  Out[m][n] = scale * sum_k A[m][k] * Bw[n][k]
//   A: f32 (A_F32, converted in-register) or bf16 (async-DMA'd); Bw: bf16 [n][k]
//   (async-DMA'd); Out: f32*scale (OUT_F32) or bf16.
//   Block tile 128(M) x 64(N), BK=64 (2 WMMA K-steps / stage), double-buffered
//   LDS with ASYNCcnt-pipelined global->LDS staging.
//   256 threads = 8 waves (4x2), each wave: 32x32 tile = 2x2 WMMA accumulators.
// ---------------------------------------------------------------------------
template <bool A_F32, bool OUT_F32>
__global__ __launch_bounds__(256) void gemm_kernel(const void* __restrict__ Ap,
                                                   const unsigned short* __restrict__ Bw,
                                                   void* __restrict__ Outp,
                                                   int M, int N, int K,
                                                   int lda, int ldb, int ldo,
                                                   const float* scale_ptr) {
    __shared__ unsigned short As[2][128 * 72];   // [m][k], stride 72
    __shared__ unsigned short Bs[2][64 * 72];    // [n][k], stride 72

    const int tid  = threadIdx.x;
    const int w    = tid >> 5;
    const int lane = tid & 31;
    const int half = lane >> 4;
    const int lr   = lane & 15;
    const int wm   = (w >> 1) * 32;
    const int wn   = (w & 1) * 32;

    const int m0 = blockIdx.y * 128;
    const int n0 = blockIdx.x * 64;

    // async instructions issued per stage (per wave): B: 2, A(bf16): 4
    constexpr int NASY = A_F32 ? 2 : 6;

    v8f acc[2][2] = {};

    const int nK = K / 64;

    // ---- staging helpers ----
    auto stage = [&](int buf, int k0) {
        // B tile (64 x 64), fully async
        {
            int row = tid >> 2, c0 = (tid & 3) * 8;
#pragma unroll
            for (int j = 0; j < 2; ++j) {
                int c8 = c0 + j * 32;
                cp_async_b128(Bw + (size_t)(n0 + row) * ldb + k0 + c8,
                              &Bs[buf][row * 72 + c8]);
            }
        }
        if constexpr (A_F32) {
            // A tile (128 x 64) f32 -> bf16 in-register, sync LDS stores
            const float* Af = (const float*)Ap;
            for (int i = tid; i < 128 * 16; i += 256) {
                int row = i >> 4, c4 = (i & 15) * 4;
                const float* gp = Af + (size_t)(m0 + row) * lda + k0 + c4;
                __builtin_prefetch((const void*)(gp + 128), 0, 1);
                v4f x = *(const v4f*)gp;
                unsigned lo = (unsigned)f2bf(x[0]) | ((unsigned)f2bf(x[1]) << 16);
                unsigned hi = (unsigned)f2bf(x[2]) | ((unsigned)f2bf(x[3]) << 16);
                *(unsigned*)&As[buf][row * 72 + c4]     = lo;
                *(unsigned*)&As[buf][row * 72 + c4 + 2] = hi;
            }
        } else {
            // A tile (128 x 64) bf16, fully async
            const unsigned short* Ah = (const unsigned short*)Ap;
            int row = tid >> 1, c0 = (tid & 1) * 8;
#pragma unroll
            for (int j = 0; j < 4; ++j) {
                int c8 = c0 + j * 16;
                cp_async_b128(Ah + (size_t)(m0 + row) * lda + k0 + c8,
                              &As[buf][row * 72 + c8]);
            }
        }
    };

    stage(0, 0);   // prologue

    for (int kt = 0; kt < nK; ++kt) {
        const int cur = kt & 1;
        const bool more = (kt + 1) < nK;
        if (more) stage(1 - cur, (kt + 1) * 64);   // prefetch next tile (DMA)

        // wait: leave only the just-issued group outstanding
        if (more) async_wait_n<NASY>(); else async_wait_n<0>();
        __syncthreads();                            // tile `cur` visible

#pragma unroll
        for (int ks = 0; ks < 2; ++ks) {
            v16bf af[2], bf2[2];
#pragma unroll
            for (int tm = 0; tm < 2; ++tm)
                af[tm] = ld_frag_a(&As[cur][0], 72, wm + tm * 16 + lr, half, ks * 32);
#pragma unroll
            for (int tn = 0; tn < 2; ++tn)
                bf2[tn] = ld_frag_b(&Bs[cur][0], 72, wn + tn * 16 + lr, half, ks * 32);
#pragma unroll
            for (int tm = 0; tm < 2; ++tm)
#pragma unroll
                for (int tn = 0; tn < 2; ++tn)
                    acc[tm][tn] = wmma_bf16(af[tm], bf2[tn], acc[tm][tn]);
        }
        __syncthreads();   // all reads of buf `cur` done before it is re-staged
    }

    const float scale = (scale_ptr != nullptr) ? *scale_ptr : 1.0f;
#pragma unroll
    for (int tm = 0; tm < 2; ++tm) {
#pragma unroll
        for (int tn = 0; tn < 2; ++tn) {
#pragma unroll
            for (int r = 0; r < 8; ++r) {
                int m = m0 + wm + tm * 16 + r + half * 8;
                int n = n0 + wn + tn * 16 + lr;
                float v = acc[tm][tn][r] * scale;
                if constexpr (OUT_F32) {
                    ((float*)Outp)[(size_t)m * ldo + n] = v;
                } else {
                    ((unsigned short*)Outp)[(size_t)m * ldo + n] = f2bf(v);
                }
            }
        }
    }
}

// ---------------------------------------------------------------------------
// Fused decayed anti-causal attention.
//   qkv: bf16 [MROWS][768]  (q|k|v each DD_DIM); ret: bf16 [MROWS][DD_DIM]
//   scores(t,s) = q(t).k(s) * decay^(s-t-1) for s>t else 0; ret = scores @ v
//   Block: 32 t-rows of one batch, 256 threads (8 waves); each wave owns 32
//   D-columns of the 32x256 accumulator (2x2 WMMA tiles). Q/K staged via
//   async-to-LDS DMA; V staged with an in-register transpose.
// ---------------------------------------------------------------------------
__global__ __launch_bounds__(256) void attn_kernel(const unsigned short* __restrict__ qkv,
                                                   unsigned short* __restrict__ ret,
                                                   const float* __restrict__ decay_ptr) {
    __shared__ unsigned short Qs[32 * 264];    // [t][d]  (A for scores)
    __shared__ unsigned short Ks[32 * 264];    // [s][d]  (B for scores, [n][k])
    __shared__ unsigned short VsT[256 * 40];   // [d][s]  (B for PV,     [n][k])
    __shared__ unsigned short Ss[32 * 40];     // [t][s]  (A for PV)

    const int b  = blockIdx.y;
    const int t0 = blockIdx.x * 32;
    const int tid  = threadIdx.x;
    const int w    = tid >> 5;
    const int lane = tid & 31;
    const int half = lane >> 4;
    const int lr   = lane & 15;
    const int nw   = w * 32;

    const float dl    = *decay_ptr;
    const float decay = 1.0f / (1.0f + __expf(-dl));
    const float ldec  = __logf(decay);

    // stage Q tile once (32 x 256), async
    {
        const unsigned short* qb = qkv + (size_t)(b * TT_DIM + t0) * NQKV;
#pragma unroll
        for (int j = 0; j < 4; ++j) {
            int i = tid + j * 256;
            int r = i >> 5, c8 = (i & 31) * 8;
            cp_async_b128(qb + (size_t)r * NQKV + c8, &Qs[r * 264 + c8]);
        }
    }

    v8f acc[2][2] = {};

    const int stEnd = TT_DIM / 32;
    for (int st = blockIdx.x; st < stEnd; ++st) {
        const int s0 = st * 32;
        __syncthreads();
        const unsigned short* kb = qkv + (size_t)(b * TT_DIM + s0) * NQKV + DD_DIM;
        const unsigned short* vb = kb + DD_DIM;
        // K tile (32 x 256) async
#pragma unroll
        for (int j = 0; j < 4; ++j) {
            int i = tid + j * 256;
            int r = i >> 5, c8 = (i & 31) * 8;
            cp_async_b128(kb + (size_t)r * NQKV + c8, &Ks[r * 264 + c8]);
        }
        // V tile transposed (256 x 32), sync
        for (int i = tid; i < 32 * 32; i += 256) {
            int r = i >> 5, c8 = (i & 31) * 8;
            v4u vx = *(const v4u*)(vb + (size_t)r * NQKV + c8);
            unsigned short tmp[8];
            *(v4u*)tmp = vx;
#pragma unroll
            for (int j = 0; j < 8; ++j) VsT[(c8 + j) * 40 + r] = tmp[j];
        }
        async_wait_n<0>();
        __syncthreads();

        // ---- scores: waves 0..3 each compute one 16x16 tile of 32x32 S ----
        if (w < 4) {
            const int tm = (w >> 1) * 16;
            const int tn = (w & 1) * 16;
            v8f sacc = {};
#pragma unroll
            for (int kk = 0; kk < DD_DIM; kk += 32) {
                v16bf a  = ld_frag_a(Qs, 264, tm + lr, half, kk);
                v16bf bb = ld_frag_b(Ks, 264, tn + lr, half, kk);
                sacc = wmma_bf16(a, bb, sacc);
            }
#pragma unroll
            for (int r = 0; r < 8; ++r) {
                int m = tm + r + half * 8;
                int n = tn + lr;
                int t = t0 + m, s = s0 + n;
                float wgt = (s > t) ? __expf(ldec * (float)(s - t - 1)) : 0.0f;
                Ss[m * 40 + n] = f2bf(sacc[r] * wgt);
            }
        }
        __syncthreads();

        // ---- acc += S @ V  (each wave: its 32 D-cols, one K-step of 32) ----
#pragma unroll
        for (int tm = 0; tm < 2; ++tm) {
            v16bf a = ld_frag_a(Ss, 40, tm * 16 + lr, half, 0);
#pragma unroll
            for (int tn = 0; tn < 2; ++tn) {
                v16bf bb = ld_frag_b(VsT, 40, nw + tn * 16 + lr, half, 0);
                acc[tm][tn] = wmma_bf16(a, bb, acc[tm][tn]);
            }
        }
    }

    unsigned short* rb = ret + (size_t)(b * TT_DIM + t0) * DD_DIM;
#pragma unroll
    for (int tm = 0; tm < 2; ++tm)
#pragma unroll
        for (int tn = 0; tn < 2; ++tn)
#pragma unroll
            for (int r = 0; r < 8; ++r) {
                int m = tm * 16 + r + half * 8;
                int n = nw + tn * 16 + lr;
                rb[(size_t)m * DD_DIM + n] = f2bf(acc[tm][tn][r]);
            }
}

// ---------------------------------------------------------------------------
// Host launcher
// ---------------------------------------------------------------------------
extern "C" void kernel_launch(void* const* d_in, const int* in_sizes, int n_in,
                              void* d_out, int out_size, void* d_ws, size_t ws_size,
                              hipStream_t stream) {
    (void)in_sizes; (void)n_in; (void)out_size; (void)ws_size;

    const float* x     = (const float*)d_in[0];   // (B,T,V)
    const float* wq    = (const float*)d_in[1];   // (D,V)
    const float* wk    = (const float*)d_in[2];
    const float* wv    = (const float*)d_in[3];
    const float* wo    = (const float*)d_in[4];   // (V,D)
    const float* dlog  = (const float*)d_in[5];
    const float* oscal = (const float*)d_in[6];

    unsigned short* ws   = (unsigned short*)d_ws;
    const size_t DV      = (size_t)DD_DIM * VV_DIM;       // 2,097,152
    unsigned short* wqkv = ws;                             // [768][8192]
    unsigned short* wob  = ws + 3 * DV;                    // [8192][256]
    unsigned short* qkv  = ws + 4 * DV;                    // [8192][768]
    unsigned short* retv = qkv + (size_t)MROWS * NQKV;     // [8192][256]

    cvt_f32_bf16<<<2048, 256, 0, stream>>>(wq, wqkv,          (int)DV);
    cvt_f32_bf16<<<2048, 256, 0, stream>>>(wk, wqkv + DV,     (int)DV);
    cvt_f32_bf16<<<2048, 256, 0, stream>>>(wv, wqkv + 2 * DV, (int)DV);
    cvt_f32_bf16<<<2048, 256, 0, stream>>>(wo, wob,           (int)DV);

    // QKV projection: [8192 x 8192] x [8192 x 768] -> bf16 qkv
    {
        dim3 grid(NQKV / 64, MROWS / 128);
        gemm_kernel<true, false><<<grid, 256, 0, stream>>>(
            x, wqkv, qkv, MROWS, NQKV, VV_DIM,
            VV_DIM, VV_DIM, NQKV, nullptr);
    }

    // fused decayed attention -> bf16 retrieved
    {
        dim3 grid(TT_DIM / 32, BB);
        attn_kernel<<<grid, 256, 0, stream>>>(qkv, retv, dlog);
    }

    // output projection: [8192 x 256] x [256 x 8192] -> f32 out * out_scale
    {
        dim3 grid(VV_DIM / 64, MROWS / 128);
        gemm_kernel<false, true><<<grid, 256, 0, stream>>>(
            retv, wob, d_out, MROWS, VV_DIM, DD_DIM,
            DD_DIM, DD_DIM, VV_DIM, oscal);
    }
}
